// QAEncoder_63316407878135
// MI455X (gfx1250) — compile-verified, matrix operand order checked by reading, not run on hardware
//
#include <hip/hip_runtime.h>
#include <hip/hip_bf16.h>

// ---------- problem constants ----------
#define BB 32
#define SS 512
#define HH 640
#define NH 10
#define DK 64
#define KK 7
#define NCONV 3
#define MROWS (BB * SS)          // 16384
#define BSH   (BB * SS * HH)     // 10485760

// ---------- vector types ----------
typedef _Float16 h8v  __attribute__((ext_vector_type(8)));
typedef _Float16 v16h __attribute__((ext_vector_type(16)));
typedef float    v8f  __attribute__((ext_vector_type(8)));
typedef unsigned int u32x4 __attribute__((ext_vector_type(4)));
typedef int          i32x8 __attribute__((ext_vector_type(8)));
typedef int          i32x4 __attribute__((ext_vector_type(4)));

#if __has_builtin(__builtin_amdgcn_tensor_load_to_lds)
#define USE_TDM 1
#else
#define USE_TDM 0
#endif

// ---------- TDM-pad address helper ----------
// TDM writes with pad_enable: 4 pad dwords inserted after every 64 data dwords.
// Effective dword address of data dword d0:  d0 + 4*floor(d0/64).
// Aligned 4-dword chunks (d0 % 4 == 0) never straddle a pad and stay 16B aligned.
__device__ __forceinline__ _Float16* lds_pad(_Float16* base, int d0) {
    int d = d0 + ((d0 >> 6) << 2);
    return base + 2 * d;
}

// ---------- WMMA fragment helpers ----------
// ISA layout (A 16x32 / B 32x16 f16, wave32): lane L, half h ->
//   K = (h<8?0:16) + ((L&16)?8:0) + (h&7)   => two contiguous 16B chunks.
__device__ __forceinline__ v16h make_frag(h8v lo, h8v hi) {
    v16h r;
#pragma unroll
    for (int i = 0; i < 8; ++i) { r[i] = lo[i]; r[8 + i] = hi[i]; }
    return r;
}
__device__ __forceinline__ v16h load_frag(const _Float16* rowBase, int kbase, int laneHi) {
    h8v lo = *(const h8v*)(rowBase + kbase + laneHi);
    h8v hi = *(const h8v*)(rowBase + kbase + 16 + laneHi);
    return make_frag(lo, hi);
}
// fragment from TDM-padded LDS panel: row stride rowDw dwords, half offset hOff
__device__ __forceinline__ v16h load_frag_pad(_Float16* base, int row, int rowDw, int hOff) {
    int d0 = row * rowDw + (hOff >> 1);
    h8v lo = *(const h8v*)lds_pad(base, d0);
    h8v hi = *(const h8v*)lds_pad(base, d0 + 8);
    return make_frag(lo, hi);
}
__device__ __forceinline__ v8f wmma16(v16h a, v16h b, v8f c) {
    return __builtin_amdgcn_wmma_f32_16x16x32_f16(false, a, false, b, (short)0, c, false, false);
}

#if USE_TDM
// ---------- Tensor Data Mover: 2D tile (nrows x rowLen halves) -> padded LDS ----------
__device__ __forceinline__ void tdm_load_2d(unsigned int lds_off, const void* gaddr,
                                            unsigned int rowLen, unsigned int nrows,
                                            unsigned int rowStride) {
    unsigned long long ga = (unsigned long long)(uintptr_t)gaddr;
    u32x4 g0;
    g0[0] = 1u;                                   // count=1, user mode
    g0[1] = lds_off;                              // lds_addr [63:32]
    g0[2] = (unsigned int)ga;                     // global_addr [95:64]
    g0[3] = (unsigned int)((ga >> 32) & 0x1FFFFFFu) | (2u << 30); // addr[56:32] | type=2
    i32x8 g1;
    // data_size=1(2B), pad_enable, pad_interval=5 (64 dw), pad_amount=3 (4 dw)
    g1[0] = (int)((1u << 16) | (1u << 20) | (5u << 22) | (3u << 25));
    g1[1] = (int)((rowLen & 0xFFFFu) << 16);                       // tensor_dim0 lo16 @ [63:48]
    g1[2] = (int)((rowLen >> 16) | ((nrows & 0xFFFFu) << 16));     // dim0 hi | dim1 lo
    g1[3] = (int)((nrows >> 16) | ((rowLen & 0xFFFFu) << 16));     // dim1 hi | tile_dim0
    g1[4] = (int)(nrows & 0xFFFFu);                                // tile_dim1 (tile_dim2=0)
    g1[5] = (int)rowStride;                                        // tensor_dim0_stride lo32
    g1[6] = 0;
    g1[7] = 0;
    i32x4 z4 = {0, 0, 0, 0};
    i32x8 z8 = {0, 0, 0, 0, 0, 0, 0, 0};
    __builtin_amdgcn_tensor_load_to_lds(g0, g1, z4, z4, z8, 0);
}
#endif

// fallback / manual stage into the same padded layout (rowLen halves per row)
__device__ __forceinline__ void stage_panel(_Float16* dst, const _Float16* src,
                                            int rowLen, int nrows, int rowStride,
                                            int tid, int nthr) {
    int segs = rowLen >> 3;                       // 16B chunks per row
    for (int i = tid; i < nrows * segs; i += nthr) {
        int row = i / segs, seg = i - row * segs;
        h8v v = *(const h8v*)(src + (size_t)row * rowStride + seg * 8);
        *(h8v*)lds_pad(dst, row * (rowLen >> 1) + seg * 4) = v;
    }
}

// ---------- float -> f16 convert ----------
__global__ void f2h_kernel(const float* __restrict__ in, _Float16* __restrict__ out, int n) {
    int i = blockIdx.x * 256 + threadIdx.x;
    if (i < n) out[i] = (_Float16)in[i];
}

// ---------- LayerNorm (one wave32 per 640-wide row), optional PE add, optional f16 mirror ----------
__global__ void ln_kernel(const float* __restrict__ in, const float* __restrict__ g,
                          const float* __restrict__ be, float* __restrict__ outf,
                          _Float16* __restrict__ outh, int addPE) {
    int lane = threadIdx.x & 31;
    int wid  = threadIdx.x >> 5;
    int row  = blockIdx.x * 8 + wid;
    const float* rp = in + (size_t)row * HH;
    int bidx = row >> 9;                          // PE bug: indexed by batch
    float vals[20];
    float s = 0.f;
#pragma unroll
    for (int i = 0; i < 20; ++i) {
        int c = i * 32 + lane;
        float v = rp[c];
        if (addPE) {
            int j = c >> 1;
            float freq = __expf((float)(2 * j) * (-9.210340371976184f / (float)HH));
            float ang  = (float)bidx * freq;
            v += (c & 1) ? __cosf(ang) : __sinf(ang);
        }
        vals[i] = v;
        s += v;
    }
#pragma unroll
    for (int off = 16; off >= 1; off >>= 1) s += __shfl_xor(s, off, 32);
    float mean = s * (1.f / (float)HH);
    float q = 0.f;
#pragma unroll
    for (int i = 0; i < 20; ++i) { float d = vals[i] - mean; q += d * d; }
#pragma unroll
    for (int off = 16; off >= 1; off >>= 1) q += __shfl_xor(q, off, 32);
    float inv = rsqrtf(q * (1.f / (float)HH) + 1e-5f);
    size_t rb = (size_t)row * HH;
#pragma unroll
    for (int i = 0; i < 20; ++i) {
        int c = i * 32 + lane;
        float o = (vals[i] - mean) * inv * g[c] + be[c];
        if (outf) outf[rb + c] = o;
        if (outh) outh[rb + c] = (_Float16)o;
    }
}

// ---------- depthwise conv along S (K=7, same pad), optional residual ----------
__global__ void dwconv_kernel(const float* __restrict__ in, const float* __restrict__ w,
                              const float* __restrict__ bias, const float* __restrict__ resid,
                              float* __restrict__ out) {
    int idx = blockIdx.x * 256 + threadIdx.x;
    int c  = idx % HH;
    int s  = (idx / HH) & (SS - 1);
    int bb = idx / (HH * SS);
    float acc = bias[c];
#pragma unroll
    for (int k = 0; k < KK; ++k) {
        int ss = s + k - (KK / 2);
        if (ss >= 0 && ss < SS)
            acc += w[c * KK + k] * in[((size_t)bb * SS + ss) * HH + c];
    }
    if (resid) acc += resid[idx];
    out[idx] = acc;
}

// ---------- WMMA GEMM with TDM-staged weight panel ----------
// block (32,8): 8 waves; workgroup tile = 128 rows x 64 cols.
// dyn LDS: padded B panel, 64 rows x 640 halves -> 87040 B
// Compile-time epilogue config: RELU, HAS_RES (f32 resid add), OUTF (f32 store), OUTH (f16 store)
template <bool RELU, bool HAS_RES, bool OUTF, bool OUTH>
__global__ void gemm_wmma_kernel(const _Float16* __restrict__ A, const _Float16* __restrict__ Bw,
                                 const float* __restrict__ bias, const float* __restrict__ resid,
                                 float* __restrict__ outf, _Float16* __restrict__ outh,
                                 int M, int N, int Kd) {
    extern __shared__ char smem[];
    _Float16* Bsh = (_Float16*)smem;
    int lane = threadIdx.x & 31;
    int w = threadIdx.y;
    int tileM = blockIdx.x * 8 + w;
    int tileN = blockIdx.y;
    int col4 = lane & 15;
    int laneHi = (lane & 16) ? 8 : 0;

    const _Float16* panel = Bw + (size_t)tileN * 64 * Kd;
#if USE_TDM
    if (w == 0) {
        tdm_load_2d((unsigned int)(uintptr_t)Bsh, panel, Kd, 64, Kd);
        __builtin_amdgcn_s_wait_tensorcnt(0);
    }
#else
    stage_panel(Bsh, panel, Kd, 64, Kd, threadIdx.y * 32 + lane, 256);
#endif
    __syncthreads();

    int row = tileM * 16 + col4;
    const _Float16* arow = A + (size_t)row * Kd;
    int rowDw = Kd >> 1;

    v8f acc[4];
#pragma unroll
    for (int nt = 0; nt < 4; ++nt)
#pragma unroll
        for (int j = 0; j < 8; ++j) acc[nt][j] = 0.f;

    for (int kb = 0; kb < Kd; kb += 32) {
        v16h a = load_frag(arow, kb, laneHi);
#pragma unroll
        for (int nt = 0; nt < 4; ++nt) {
            v16h b = load_frag_pad(Bsh, nt * 16 + col4, rowDw, kb + laneHi);
            acc[nt] = wmma16(a, b, acc[nt]);
        }
    }
    // C layout: elem j of lane L -> row j + ((L&16)?8:0), col L&15
#pragma unroll
    for (int nt = 0; nt < 4; ++nt) {
        int gcol = tileN * 64 + nt * 16 + col4;
        float bv = bias[gcol];
#pragma unroll
        for (int j = 0; j < 8; ++j) {
            int gm = tileM * 16 + laneHi + j;
            float v = acc[nt][j] + bv;
            if (RELU) v = v > 0.f ? v : 0.f;
            size_t off = (size_t)gm * N + gcol;
            if (HAS_RES) v += resid[off];
            if (OUTF) outf[off] = v;
            if (OUTH) outh[off] = (_Float16)v;
        }
    }
}

// ---------- flash attention: grid (S/128, NH, B), block 256 (8 waves) ----------
// dyn LDS: Ksh padded (512x64h, TDM)   = 69632 B
//          VshT (64 x 520h, stride-padded transpose) = 66560 B
//          Psh  (8 waves x 16 x 40h)   = 10240 B      total 146432 B
#define VST 520
#define PST 40
__global__ void attn_kernel(const _Float16* __restrict__ Q, const _Float16* __restrict__ K,
                            const _Float16* __restrict__ V, _Float16* __restrict__ O) {
    extern __shared__ char smem[];
    _Float16* Ksh  = (_Float16*)smem;                 // TDM-padded, 32 dw/row
    _Float16* VshT = (_Float16*)(smem + 69632);
    _Float16* Psh  = (_Float16*)(smem + 69632 + 66560);

    int tid = threadIdx.x;
    int lane = tid & 31, w = tid >> 5;
    int b = blockIdx.z, h = blockIdx.y;
    int qb = blockIdx.x * 128 + w * 16;
    const size_t base = ((size_t)b * SS) * HH + (size_t)h * DK;

#if USE_TDM
    if (w == 0) {
        tdm_load_2d((unsigned int)(uintptr_t)Ksh, K + base, DK, SS, HH);
        __builtin_amdgcn_s_wait_tensorcnt(0);
    }
#else
    stage_panel(Ksh, K + base, DK, SS, HH, tid, 256);
#endif
    // transpose-stage V: VshT[d*VST + key]
    for (int i = tid; i < SS * 8; i += 256) {
        int key = i >> 3, seg = i & 7;
        h8v vv = *(const h8v*)(V + base + (size_t)key * HH + seg * 8);
#pragma unroll
        for (int t = 0; t < 8; ++t) VshT[(seg * 8 + t) * VST + key] = vv[t];
    }
    __syncthreads();

    int col4 = lane & 15;
    int laneHi = (lane & 16) ? 8 : 0;
    const _Float16* qrow = Q + base + (size_t)(qb + col4) * HH;
    v16h aq0 = load_frag(qrow, 0, laneHi);
    v16h aq1 = load_frag(qrow, 32, laneHi);

    float mrun[8], lrun[8];
    v8f o[4];
#pragma unroll
    for (int j = 0; j < 8; ++j) { mrun[j] = -1e30f; lrun[j] = 0.f; }
#pragma unroll
    for (int nt = 0; nt < 4; ++nt)
#pragma unroll
        for (int j = 0; j < 8; ++j) o[nt][j] = 0.f;

    _Float16* pw = Psh + w * (16 * PST);

    for (int kt = 0; kt < SS / 32; ++kt) {
        int kb = kt * 32;
        v8f s0, s1;
#pragma unroll
        for (int j = 0; j < 8; ++j) { s0[j] = 0.f; s1[j] = 0.f; }
        // keys [kb, kb+16)
        s0 = wmma16(aq0, load_frag_pad(Ksh, kb + col4, 32, laneHi), s0);
        s0 = wmma16(aq1, load_frag_pad(Ksh, kb + col4, 32, 32 + laneHi), s0);
        // keys [kb+16, kb+32)
        s1 = wmma16(aq0, load_frag_pad(Ksh, kb + 16 + col4, 32, laneHi), s1);
        s1 = wmma16(aq1, load_frag_pad(Ksh, kb + 16 + col4, 32, 32 + laneHi), s1);

        // online softmax over this 32-key chunk; rows live in 16-lane halves
#pragma unroll
        for (int j = 0; j < 8; ++j) {
            float a0 = s0[j] * 0.125f;            // 1/sqrt(64)
            float a1 = s1[j] * 0.125f;
            float rm = fmaxf(a0, a1);
#pragma unroll
            for (int off = 8; off >= 1; off >>= 1) rm = fmaxf(rm, __shfl_xor(rm, off, 32));
            float mnew = fmaxf(mrun[j], rm);
            float corr = __expf(mrun[j] - mnew);
            mrun[j] = mnew;
            float p0 = __expf(a0 - mnew), p1 = __expf(a1 - mnew);
            float rs = p0 + p1;
#pragma unroll
            for (int off = 8; off >= 1; off >>= 1) rs += __shfl_xor(rs, off, 32);
            lrun[j] = lrun[j] * corr + rs;
#pragma unroll
            for (int nt = 0; nt < 4; ++nt) o[nt][j] *= corr;
            int rloc = j + laneHi;
            pw[rloc * PST + col4]      = (_Float16)p0;
            pw[rloc * PST + 16 + col4] = (_Float16)p1;
        }
        // re-layout P (D-layout -> A fragment) via per-wave LDS patch
        v16h ap = load_frag(pw + col4 * PST, 0, laneHi);
#pragma unroll
        for (int nt = 0; nt < 4; ++nt) {
            const _Float16* vrow = VshT + (nt * 16 + col4) * VST + kb;
            o[nt] = wmma16(ap, load_frag(vrow, 0, laneHi), o[nt]);
        }
    }
#pragma unroll
    for (int nt = 0; nt < 4; ++nt)
#pragma unroll
        for (int j = 0; j < 8; ++j) {
            int gm = qb + laneHi + j;
            float val = o[nt][j] / lrun[j];
            O[((size_t)b * SS + gm) * HH + h * DK + nt * 16 + col4] = (_Float16)val;
        }
}

// ===================== host launcher =====================
extern "C" void kernel_launch(void* const* d_in, const int* in_sizes, int n_in,
                              void* d_out, int out_size, void* d_ws, size_t ws_size,
                              hipStream_t stream) {
    const float* x_in       = (const float*)d_in[0];
    const float* init_ln_g  = (const float*)d_in[1];
    const float* init_ln_b  = (const float*)d_in[2];
    const float* ln_g       = (const float*)d_in[3];
    const float* ln_b       = (const float*)d_in[4];
    const float* init_cw    = (const float*)d_in[5];
    const float* init_cb    = (const float*)d_in[6];
    const float* conv_w     = (const float*)d_in[7];
    const float* conv_b     = (const float*)d_in[8];
    const float* wq = (const float*)d_in[9];   const float* bq = (const float*)d_in[10];
    const float* wk = (const float*)d_in[11];  const float* bk = (const float*)d_in[12];
    const float* wv = (const float*)d_in[13];  const float* bv = (const float*)d_in[14];
    const float* wo = (const float*)d_in[15];  const float* bo = (const float*)d_in[16];
    const float* ffw = (const float*)d_in[17]; const float* ffb = (const float*)d_in[18];

    float* X = (float*)d_out;                    // resident f32 activation == final output

    char* ws = (char*)d_ws;
    float*    t    = (float*)ws;      ws += (size_t)BSH * 4;
    _Float16* xn   = (_Float16*)ws;   ws += (size_t)BSH * 2;
    _Float16* qh   = (_Float16*)ws;   ws += (size_t)BSH * 2;
    _Float16* kh   = (_Float16*)ws;   ws += (size_t)BSH * 2;
    _Float16* vh   = (_Float16*)ws;   ws += (size_t)BSH * 2;
    _Float16* ctxh = (_Float16*)ws;   ws += (size_t)BSH * 2;
    _Float16* whq  = (_Float16*)ws;   ws += (size_t)HH * HH * 2;
    _Float16* whk  = (_Float16*)ws;   ws += (size_t)HH * HH * 2;
    _Float16* whv  = (_Float16*)ws;   ws += (size_t)HH * HH * 2;
    _Float16* who  = (_Float16*)ws;   ws += (size_t)HH * HH * 2;
    _Float16* whf  = (_Float16*)ws;   ws += (size_t)HH * HH * 2;

    dim3 lnGrid(MROWS / 8), lnBlk(256);
    dim3 cvGrid(BSH / 256), cvBlk(256);
    dim3 gGrid(MROWS / 128, HH / 64), gBlk(32, 8);
    size_t gSmem = 87040;                        // padded 64 x 640 f16 panel

    // 1) x + PE (batch-indexed) -> init LayerNorm -> t
    ln_kernel<<<lnGrid, lnBlk, 0, stream>>>(x_in, init_ln_g, init_ln_b, t, nullptr, 1);
    // 2) init depthwise conv: X = conv(t)
    dwconv_kernel<<<cvGrid, cvBlk, 0, stream>>>(t, init_cw, init_cb, nullptr, X);
    // 3) residual conv stack
    for (int i = 0; i < NCONV; ++i) {
        ln_kernel<<<lnGrid, lnBlk, 0, stream>>>(X, ln_g, ln_b, t, nullptr, 0);
        dwconv_kernel<<<cvGrid, cvBlk, 0, stream>>>(t, conv_w + (size_t)i * HH * KK,
                                                    conv_b + (size_t)i * HH, X, X);
    }
    // 4) attention pre-LN (f16 mirror for GEMMs)
    ln_kernel<<<lnGrid, lnBlk, 0, stream>>>(X, ln_g, ln_b, t, xn, 0);
    // 5) weights -> f16
    {
        int n = HH * HH;
        dim3 wg((n + 255) / 256), wb(256);
        f2h_kernel<<<wg, wb, 0, stream>>>(wq,  whq, n);
        f2h_kernel<<<wg, wb, 0, stream>>>(wk,  whk, n);
        f2h_kernel<<<wg, wb, 0, stream>>>(wv,  whv, n);
        f2h_kernel<<<wg, wb, 0, stream>>>(wo,  who, n);
        f2h_kernel<<<wg, wb, 0, stream>>>(ffw, whf, n);
    }
    // 6) Q/K/V projections (f16 outputs):       RELU=0, RES=0, OUTF=0, OUTH=1
    gemm_wmma_kernel<false, false, false, true><<<gGrid, gBlk, gSmem, stream>>>(
        xn, whq, bq, nullptr, nullptr, qh, MROWS, HH, HH);
    gemm_wmma_kernel<false, false, false, true><<<gGrid, gBlk, gSmem, stream>>>(
        xn, whk, bk, nullptr, nullptr, kh, MROWS, HH, HH);
    gemm_wmma_kernel<false, false, false, true><<<gGrid, gBlk, gSmem, stream>>>(
        xn, whv, bv, nullptr, nullptr, vh, MROWS, HH, HH);
    // 7) flash attention -> ctxh
    {
        dim3 aGrid(SS / 128, NH, BB), aBlk(256);
        size_t smemA = 69632 + 66560 + 10240;    // 146432 B
        attn_kernel<<<aGrid, aBlk, smemA, stream>>>(qh, kh, vh, ctxh);
    }
    // 8) output projection + residual: X = ctx @ wo^T + bo + X   (RES=1, OUTF=1)
    gemm_wmma_kernel<false, true, true, false><<<gGrid, gBlk, gSmem, stream>>>(
        ctxh, who, bo, X, X, nullptr, MROWS, HH, HH);
    // 9) FFN pre-LN
    ln_kernel<<<lnGrid, lnBlk, 0, stream>>>(X, ln_g, ln_b, t, xn, 0);
    // 10) FFN: X = relu(xn @ ffw^T + ffb) + X   (RELU=1, RES=1, OUTF=1)
    gemm_wmma_kernel<true, true, true, false><<<gGrid, gBlk, gSmem, stream>>>(
        xn, whf, ffb, X, X, nullptr, MROWS, HH, HH);
}
